// SNColBERTSim_55662776156185
// MI455X (gfx1250) — matrix operand order repaired and checked by reading, not compiled, not run on hardware
//
#include <hip/hip_runtime.h>
#include <hip/hip_bf16.h>
#include <stdint.h>

typedef _Float16 v16h __attribute__((ext_vector_type(16)));
typedef _Float16 v8h  __attribute__((ext_vector_type(8)));
typedef float    v8f  __attribute__((ext_vector_type(8)));

#define NEGV (-99999.0f)
#define D 128

// ---------------------------------------------------------------------------
// Kernel 1: fused score -> exact top-k (jax tie-break) -> sorted gather (f32->f16)
// One block per (b,n) row; blockDim.x == L tokens.
// ---------------------------------------------------------------------------
__global__ void score_topk_gather(const float* __restrict__ rep,
                                  const unsigned char* __restrict__ mask,
                                  const float* __restrict__ W,
                                  const float* __restrict__ bias,
                                  _Float16* __restrict__ gOut,
                                  unsigned char* __restrict__ mOut,
                                  int L, int K) {
  __shared__ float sW[D];
  __shared__ float sS[256];
  __shared__ unsigned char sF[256];

  const int row = blockIdx.x;
  const int l   = threadIdx.x;

  if (l < D) sW[l] = W[l];
  __syncthreads();

  const float* r = rep + ((size_t)row * L + l) * D;
  float acc = 0.0f;
#pragma unroll
  for (int d = 0; d < D; d += 4) {
    float4 v = *(const float4*)(r + d);
    acc += v.x * sW[d] + v.y * sW[d + 1] + v.z * sW[d + 2] + v.w * sW[d + 3];
  }
  acc += bias[0];

  const unsigned char mk = mask[(size_t)row * L + l];
  const float sc = mk ? acc : NEGV;
  sS[l] = sc;
  __syncthreads();

  // rank = number of tokens strictly better (ties broken by lower index)
  int rank = 0;
  for (int j = 0; j < L; ++j) {
    const float sj = sS[j];
    rank += (sj > sc) || (sj == sc && j < l);
  }
  const int sel = (rank < K) ? 1 : 0;
  sF[l] = (unsigned char)sel;
  __syncthreads();

  if (sel) {
    int pos = 0;
    for (int j = 0; j < l; ++j) pos += sF[j];   // compact in index order == sorted idx
    _Float16* dst = gOut + ((size_t)row * K + pos) * D;
#pragma unroll
    for (int d = 0; d < D; d += 8) {
      float4 a = *(const float4*)(r + d);
      float4 c = *(const float4*)(r + d + 4);
      v8h h;
      h[0] = (_Float16)a.x; h[1] = (_Float16)a.y; h[2] = (_Float16)a.z; h[3] = (_Float16)a.w;
      h[4] = (_Float16)c.x; h[5] = (_Float16)c.y; h[6] = (_Float16)c.z; h[7] = (_Float16)c.w;
      *(v8h*)(dst + d) = h;
    }
    mOut[(size_t)row * K + pos] = mk;
  }
}

// ---------------------------------------------------------------------------
// Kernel 2: per (b,q,kc) block: S = A(64x128) * B(128x128)^T via WMMA f16,
// max over cand tokens (M), masked mean over ctxt tokens (N).
// 8 waves; wave w owns N-columns [16w,16w+16), loops 4 M-tiles, 4 K-steps.
// ---------------------------------------------------------------------------
#define LDA 136   // 128 + 8 halves pad -> 272B row stride, spreads LDS banks
#define LDB 136

typedef union { v16h v; v8h h[2]; } v16u;

__global__ __launch_bounds__(256) void maxsim_wmma(const _Float16* __restrict__ candG,
                                                   const _Float16* __restrict__ ctxtG,
                                                   const unsigned char* __restrict__ mAg,
                                                   const unsigned char* __restrict__ mCg,
                                                   float* __restrict__ out,
                                                   int NQ, int NK) {
  __shared__ _Float16 sA[64 * LDA];
  __shared__ _Float16 sB[128 * LDB];
  __shared__ unsigned char sMA[64];
  __shared__ unsigned char sMC[128];
  __shared__ float sSum[8];
  __shared__ float sDen[8];

  const int bid = blockIdx.x;
  const int kc  = bid % NK;
  const int q   = (bid / NK) % NQ;
  const int b   = bid / (NK * NQ);
  const int tid = threadIdx.x;

  // Stage tiles into LDS (u32 granularity; dest rows padded)
  const uint32_t* gA = (const uint32_t*)(candG + ((size_t)(b * NQ + q) * 64) * D);
  const uint32_t* gB = (const uint32_t*)(ctxtG + ((size_t)(b * NK + kc) * 128) * D);
  uint32_t* uA = (uint32_t*)sA;
  uint32_t* uB = (uint32_t*)sB;
  for (int i = tid; i < 64 * 64; i += 256) {
    const int rr = i >> 6, cc = i & 63;
    uA[rr * (LDA / 2) + cc] = gA[rr * 64 + cc];
  }
  for (int i = tid; i < 128 * 64; i += 256) {
    const int rr = i >> 6, cc = i & 63;
    uB[rr * (LDB / 2) + cc] = gB[rr * 64 + cc];
  }
  if (tid < 64)  sMA[tid] = mAg[(size_t)(b * NQ + q) * 64 + tid];
  if (tid < 128) sMC[tid] = mCg[(size_t)(b * NK + kc) * 128 + tid];
  __syncthreads();

  const int lane = tid & 31;
  const int w    = tid >> 5;       // wave id 0..7 -> N-tile
  const int lh   = lane & 15;
  const int hi   = lane >> 4;      // 0: low half, 1: high half of wave

  v8f acc0 = {}, acc1 = {}, acc2 = {}, acc3 = {};
  const _Float16* brow = sB + (w * 16 + lh) * LDB;   // B col N = 16w + lh

#pragma unroll
  for (int ks = 0; ks < 4; ++ks) {
    // B fragment (32x16, f16): lane<16 -> K={0..7,16..23}, lane>=16 -> K={8..15,24..31}
    const int db = ks * 32 + hi * 8;
    v16u bf;
    bf.h[0] = *(const v8h*)(brow + db);
    bf.h[1] = *(const v8h*)(brow + db + 16);

    // A fragments (16x32, f16): lane<16 -> K={0..7,16..23}, lane>=16 -> K={8..15,24..31}
    const int da = ks * 32 + hi * 8;
#pragma unroll
    for (int mt = 0; mt < 4; ++mt) {
      const _Float16* arow = sA + (mt * 16 + lh) * LDA;
      v16u af;
      af.h[0] = *(const v8h*)(arow + da);
      af.h[1] = *(const v8h*)(arow + da + 16);
      v8f& acc = (mt == 0) ? acc0 : (mt == 1) ? acc1 : (mt == 2) ? acc2 : acc3;
      acc = __builtin_amdgcn_wmma_f32_16x16x32_f16(
          /*neg_a=*/false, af.v, /*neg_b=*/false, bf.v,
          /*c_mod=*/(short)0, acc, /*reuse_a=*/false, /*reuse_b=*/false);
    }
  }

  // MaxSim over cand tokens (M), honoring gathered cand mask.
  // C layout: acc[r] = S[m = mt*16 + 8*hi + r][e = 16w + lh]
  float cmax = NEGV;
#pragma unroll
  for (int mt = 0; mt < 4; ++mt) {
    const v8f& acc = (mt == 0) ? acc0 : (mt == 1) ? acc1 : (mt == 2) ? acc2 : acc3;
#pragma unroll
    for (int rr = 0; rr < 8; ++rr) {
      const int m = mt * 16 + hi * 8 + rr;
      const float vv = acc[rr];
      cmax = sMA[m] ? fmaxf(cmax, vv) : cmax;
    }
  }
  cmax = fmaxf(cmax, __shfl_xor(cmax, 16));  // merge M halves (same column e)

  // Masked sum over ctxt tokens for this wave's 16 columns
  const int e = w * 16 + lh;
  const float mc = sMC[e] ? 1.0f : 0.0f;
  float vsum = cmax * mc;
  float dsum = mc;
#pragma unroll
  for (int off = 1; off < 16; off <<= 1) {
    vsum += __shfl_xor(vsum, off);
    dsum += __shfl_xor(dsum, off);
  }
  if (lane == 0) { sSum[w] = vsum; sDen[w] = dsum; }
  __syncthreads();

  if (tid == 0) {
    float t = 0.0f, d = 0.0f;
#pragma unroll
    for (int i = 0; i < 8; ++i) { t += sSum[i]; d += sDen[i]; }
    out[bid] = t / d;   // bid == (b*NQ + q)*NK + kc
  }
}

// ---------------------------------------------------------------------------
// Launch: B=8, NQ=32, LQ=128 (k=64), NK=16, LK=256 (k=128), D=128
// ws layout: candG f16 4MB | ctxtG f16 4MB | candMask 16KB | ctxtMask 16KB
// ---------------------------------------------------------------------------
extern "C" void kernel_launch(void* const* d_in, const int* in_sizes, int n_in,
                              void* d_out, int out_size, void* d_ws, size_t ws_size,
                              hipStream_t stream) {
  (void)in_sizes; (void)n_in; (void)out_size; (void)ws_size;
  const float* cand          = (const float*)d_in[0];
  const float* ctxt          = (const float*)d_in[1];
  const unsigned char* mcand = (const unsigned char*)d_in[2];
  const unsigned char* mctxt = (const unsigned char*)d_in[3];
  const float* W             = (const float*)d_in[4];
  const float* bias          = (const float*)d_in[5];
  float* out = (float*)d_out;

  char* ws = (char*)d_ws;
  _Float16* candG = (_Float16*)ws;                               // 8*32*64*128*2  = 4 MB
  _Float16* ctxtG = (_Float16*)(ws + 4194304);                   // 8*16*128*128*2 = 4 MB
  unsigned char* mAG = (unsigned char*)(ws + 8388608);           // 16 KB
  unsigned char* mCG = (unsigned char*)(ws + 8388608 + 16384);   // 16 KB

  const int B = 8, NQ = 32, NK = 16;

  // cand: 256 rows, L=128, k=64
  score_topk_gather<<<B * NQ, 128, 0, stream>>>(cand, mcand, W, bias, candG, mAG, 128, 64);
  // ctxt: 128 rows, L=256, k=128
  score_topk_gather<<<B * NK, 256, 0, stream>>>(ctxt, mctxt, W, bias, ctxtG, mCG, 256, 128);
  // 4096 blocks, one per (b,q,kc)
  maxsim_wmma<<<B * NQ * NK, 256, 0, stream>>>(candG, ctxtG, mAG, mCG, out, NQ, NK);
}
